// MAB_63496796504418
// MI455X (gfx1250) — compile-verified, matrix operand order checked by reading, not run on hardware
//
#include <hip/hip_runtime.h>
#include <hip/hip_bf16.h>

typedef __attribute__((ext_vector_type(16))) _Float16 v16h;
typedef __attribute__((ext_vector_type(8)))  _Float16 v8h;
typedef __attribute__((ext_vector_type(8)))  float    v8f;

#define DD 128
#define DH 32
#define HH 4
#define BBATCH 4
#define NN 2048
#define MMKV 2048
#define LDT 136   // padded f16 stride for 128-wide LDS tiles (16B-aligned rows, 4-bank advance)
#define LDK 40    // padded f16 stride for 32-wide LDS tiles (16B-aligned rows, 20-dword advance)

__device__ __forceinline__ v8f wmma_f16(v16h a, v16h b, v8f c) {
  // D = A(16x32 f16) * B(32x16 f16) + C(16x16 f32)
  return __builtin_amdgcn_wmma_f32_16x16x32_f16(false, a, false, b, (short)0, c,
                                                false, false);
}

// Two contiguous 16B LDS reads -> one 16-wide f16 fragment (ds_load_b128 x2).
__device__ __forceinline__ v16h load_frag16(const _Float16* p) {
  v8h lo = *(const v8h*)(p);
  v8h hi = *(const v8h*)(p + 16);
  return __builtin_shufflevector(lo, hi, 0, 1, 2, 3, 4, 5, 6, 7,
                                 8, 9, 10, 11, 12, 13, 14, 15);
}

// A-matrix fragment (16x32 f16), ISA 7.12.2 layout. Also used for B-fragments
// when the B operand is staged TRANSPOSED in LDS.
__device__ __forceinline__ v16h load_afrag(const _Float16* base, int ld) {
  const int lane = threadIdx.x & 31;
  const int row  = lane & 15;
  const int kb   = (lane < 16) ? 0 : 8;
  return load_frag16(base + row * ld + kb);
}

// CDNA5 async DMA: global -> LDS, 16B per lane, tracked by ASYNCcnt.
__device__ __forceinline__ void async_load_b128(void* lds, const void* g) {
  unsigned l = (unsigned)(uintptr_t)lds;               // LDS aperture low bits
  unsigned long long a = (unsigned long long)(uintptr_t)g;
  asm volatile("global_load_async_to_lds_b128 %0, %1, off"
               :: "v"(l), "v"(a) : "memory");
}
__device__ __forceinline__ void wait_async() {
  asm volatile("s_wait_asynccnt 0x0" ::: "memory");
}

// DPP butterfly reductions across a 16-lane row (no DS traffic, no waits).
// dpp_ctrl must be a compile-time constant -> template parameter.
template <int CTRL>
__device__ __forceinline__ float dpp_xfer(float x) {
  return __int_as_float(__builtin_amdgcn_update_dpp(
      0, __float_as_int(x), CTRL, 0xf, 0xf, true));
}
__device__ __forceinline__ float red_max16(float x) {
  x = fmaxf(x, dpp_xfer<0xB1>(x));   // quad_perm(1,0,3,2)  : xor 1
  x = fmaxf(x, dpp_xfer<0x4E>(x));   // quad_perm(2,3,0,1)  : xor 2
  x = fmaxf(x, dpp_xfer<0x141>(x));  // row_half_mirror     : combine quads
  x = fmaxf(x, dpp_xfer<0x140>(x));  // row_mirror          : combine octets
  return x;
}
__device__ __forceinline__ float red_sum16(float x) {
  x += dpp_xfer<0xB1>(x);
  x += dpp_xfer<0x4E>(x);
  x += dpp_xfer<0x141>(x);
  x += dpp_xfer<0x140>(x);
  return x;
}

// ---------------------------------------------------------------------------
// Kernel 1: Y = X @ W + bias for a 128-row tile, K = 128.
// W staged TRANSPOSED in LDS so B-fragments are contiguous b128 gathers.
// WRITE_F32 is a compile-time flag -> branch-free store loop for K/V.
// ---------------------------------------------------------------------------
template <bool WRITE_F32>
__global__ void proj_kernel(const float* __restrict__ X, const float* __restrict__ W,
                            const float* __restrict__ bias,
                            float* __restrict__ Yf, _Float16* __restrict__ Yh) {
  extern __shared__ char smem[];
  _Float16* sx  = (_Float16*)smem;      // [128][LDT] row-major X tile
  _Float16* swT = sx + DD * LDT;        // [128][LDT] W transposed: swT[n][k]
  const int t = threadIdx.x;
  const int rowbase = blockIdx.x * 128;

#pragma unroll
  for (int i = 0; i < 32; ++i) {
    const int idx = i * 512 + t * 4;
    const int r = idx >> 7, c = idx & 127;
    float4 vx = *(const float4*)(X + (size_t)(rowbase + r) * DD + c);
    sx[r * LDT + c + 0] = (_Float16)vx.x;
    sx[r * LDT + c + 1] = (_Float16)vx.y;
    sx[r * LDT + c + 2] = (_Float16)vx.z;
    sx[r * LDT + c + 3] = (_Float16)vx.w;
    float4 vw = *(const float4*)(W + idx);       // W[r=k][c=n]
    swT[(c + 0) * LDT + r] = (_Float16)vw.x;     // transpose into swT[n][k]
    swT[(c + 1) * LDT + r] = (_Float16)vw.y;
    swT[(c + 2) * LDT + r] = (_Float16)vw.z;
    swT[(c + 3) * LDT + r] = (_Float16)vw.w;
  }
  __syncthreads();

  const int wv = t >> 5, lane = t & 31;
  const int col = lane & 15;
  const int rb  = (lane < 16) ? 0 : 8;
  const int r0  = wv * 32;

  for (int cf = 0; cf < 8; ++cf) {
    v8f acc0 = {0.f,0.f,0.f,0.f,0.f,0.f,0.f,0.f};
    v8f acc1 = {0.f,0.f,0.f,0.f,0.f,0.f,0.f,0.f};
#pragma unroll
    for (int k = 0; k < 4; ++k) {
      v16h a0 = load_afrag(sx + r0 * LDT + k * 32, LDT);
      v16h a1 = load_afrag(sx + (r0 + 16) * LDT + k * 32, LDT);
      v16h bb = load_afrag(swT + (cf * 16) * LDT + k * 32, LDT);  // B via transpose
      acc0 = wmma_f16(a0, bb, acc0);
      acc1 = wmma_f16(a1, bb, acc1);
    }
    const float bv = bias[cf * 16 + col];
#pragma unroll
    for (int v = 0; v < 8; ++v) {
      const int gr0 = rowbase + r0 + rb + v;
      const int gr1 = gr0 + 16;
      const int gc  = cf * 16 + col;
      const float y0 = acc0[v] + bv;
      const float y1 = acc1[v] + bv;
      if (WRITE_F32) {
        Yf[(size_t)gr0 * DD + gc] = y0;
        Yf[(size_t)gr1 * DD + gc] = y1;
      }
      Yh[(size_t)gr0 * DD + gc] = (_Float16)y0;
      Yh[(size_t)gr1 * DD + gc] = (_Float16)y1;
    }
  }
}

// ---------------------------------------------------------------------------
// Kernel 2: fused masked flash-attention per (head, batch, 64 query rows).
// Block: 128 threads = 4 waves, each wave owns 16 query rows.
// K + mask tiles streamed via GLOBAL_LOAD_ASYNC_TO_LDS (ASYNCcnt);
// V tile transposed through VGPRs so P@V B-fragments are b128 gathers.
// Softmax reductions via DPP butterflies (16-lane rows).
// ---------------------------------------------------------------------------
__global__ void attn_kernel(const _Float16* __restrict__ Qh,
                            const _Float16* __restrict__ Kh,
                            const _Float16* __restrict__ Vh,
                            const float* __restrict__ Qf,
                            const int* __restrict__ mask,
                            float* __restrict__ O) {
  extern __shared__ char smem[];
  int*      sM  = (int*)smem;                 // [64][32] mask tile
  _Float16* sK  = (_Float16*)(sM + 64 * 32);  // [32 kv][LDK] row-major
  _Float16* sVT = sK + 32 * LDK;              // [32 dh][LDK] V transposed
  _Float16* sP  = sVT + 32 * LDK;             // [4][16][LDK] per-wave P tiles

  const int t = threadIdx.x;
  const int wv = t >> 5, lane = t & 31;
  const int b = blockIdx.y, h = blockIdx.z;
  const int q64 = blockIdx.x * 64;
  const int qb  = q64 + wv * 16;
  const int mb  = (h * BBATCH + b) / HH;   // faithful repeat_interleave indexing
  const int col = lane & 15;
  const int rb  = (lane < 16) ? 0 : 8;

  // Q A-fragment (per-wave), pre-scaled by 1/sqrt(dh)
  v16h qa;
  {
    const _Float16* qbase = Qh + ((size_t)(b * NN) + qb) * DD + h * DH;
    const int kb = (lane < 16) ? 0 : 8;
    v16h q = load_frag16(qbase + (lane & 15) * DD + kb);
    const _Float16 sc = (_Float16)0.17677669529663687f;
#pragma unroll
    for (int i = 0; i < 16; ++i) qa[i] = q[i] * sc;
  }

  float mrow[8], lrow[8];
#pragma unroll
  for (int v = 0; v < 8; ++v) { mrow[v] = -1e30f; lrow[v] = 0.f; }
  v8f o0 = {0.f,0.f,0.f,0.f,0.f,0.f,0.f,0.f};
  v8f o1 = {0.f,0.f,0.f,0.f,0.f,0.f,0.f,0.f};

  // per-thread staging indices
  const int rkv = t >> 2, ckv = (t & 3) * 8;       // K/V tile: 32 x 32, 8 f16/thread
  const int rmk = t >> 1, cmk = (t & 1) * 16;      // mask tile: 64 x 32, 16 i32/thread

  for (int mt = 0; mt < MMKV; mt += 32) {
    __syncthreads();   // previous iteration's LDS reads complete
    const size_t gk = ((size_t)(b * MMKV) + mt + rkv) * DD + h * DH + ckv;
    // K tile: async DMA straight into LDS (no VGPR round-trip)
    async_load_b128(sK + rkv * LDK + ckv, Kh + gk);
    // mask tile: 64B per thread via 4 async b128 transfers
    const int* mp = mask + ((size_t)(mb * NN) + q64 + rmk) * MMKV + mt + cmk;
#pragma unroll
    for (int i = 0; i < 4; ++i)
      async_load_b128(sM + rmk * 32 + cmk + i * 4, mp + i * 4);
    if (mt + 32 < MMKV) __builtin_prefetch(mp + 32, 0, 1);  // next mask tile
    // V tile: load + transpose through VGPRs (needed for B-fragment layout)
    {
      v8h vvv = *(const v8h*)(Vh + gk);
#pragma unroll
      for (int j = 0; j < 8; ++j) sVT[(ckv + j) * LDK + rkv] = vvv[j];
    }
    wait_async();
    __syncthreads();

    // S = Q * K^T : K^T B-fragments are A-style gathers on the row-major K tile
    const v8f z = {0.f,0.f,0.f,0.f,0.f,0.f,0.f,0.f};
    v16h bk0 = load_afrag(sK, LDK);
    v16h bk1 = load_afrag(sK + 16 * LDK, LDK);
    v8f s0 = wmma_f16(qa, bk0, z);
    v8f s1 = wmma_f16(qa, bk1, z);

    // mask + online softmax (rows live in lane-halves; DPP reduce across 16 lanes)
#pragma unroll
    for (int v = 0; v < 8; ++v) {
      const int rr = rb + v;
      const int mi = (wv * 16 + rr) * 32;
      float x0 = (sM[mi + col] == 0)      ? -1e30f : s0[v];
      float x1 = (sM[mi + 16 + col] == 0) ? -1e30f : s1[v];
      const float mx = red_max16(fmaxf(x0, x1));
      const float mnew = fmaxf(mrow[v], mx);
      const float sc = __expf(mrow[v] - mnew);
      const float p0 = __expf(x0 - mnew);
      const float p1 = __expf(x1 - mnew);
      const float ps = red_sum16(p0 + p1);
      lrow[v] = lrow[v] * sc + ps;
      mrow[v] = mnew;
      o0[v] *= sc;
      o1[v] *= sc;
      _Float16* pp = sP + (wv * 16 + rr) * LDK;
      pp[col]      = (_Float16)p0;
      pp[16 + col] = (_Float16)p1;
    }
    // intra-wave cross-lane RAW through LDS: drain DS before re-reading P
    asm volatile("s_wait_dscnt 0x0" ::: "memory");

    v16h pa  = load_afrag(sP + wv * 16 * LDK, LDK);
    v16h bv0 = load_afrag(sVT, LDK);             // B via transposed V tile
    v16h bv1 = load_afrag(sVT + 16 * LDK, LDK);
    o0 = wmma_f16(pa, bv0, o0);
    o1 = wmma_f16(pa, bv1, o1);
  }

  // O = Qh + (A @ Vh); write head h into columns [h*32, h*32+32) of [B,N,128]
#pragma unroll
  for (int v = 0; v < 8; ++v) {
    const int gr = qb + rb + v;
    const float inv = 1.f / lrow[v];
    const size_t base = ((size_t)(b * NN) + gr) * DD + h * DH;
    O[base + col]      = Qf[base + col]      + o0[v] * inv;
    O[base + 16 + col] = Qf[base + 16 + col] + o1[v] * inv;
  }
}

// ---------------------------------------------------------------------------
// Kernel 3: LN0 -> relu(X@W1+b1)@W2+b2 residual -> LN1, per 64-row tile.
// Block: 128 threads (4 waves, 16 rows each). Dynamic LDS: ~100 KB (<320 KB WGP).
// ---------------------------------------------------------------------------
__global__ void epilogue_kernel(const float* __restrict__ O,
                                const float* __restrict__ W1, const float* __restrict__ b1,
                                const float* __restrict__ W2, const float* __restrict__ b2,
                                const float* __restrict__ g0, const float* __restrict__ be0,
                                const float* __restrict__ g1, const float* __restrict__ be1,
                                float* __restrict__ out) {
  extern __shared__ char smem[];
  float*    sO  = (float*)smem;              // [64][128] f32
  _Float16* sx  = (_Float16*)(sO + 64 * DD); // [64][LDT] f16 (LN0 out)
  _Float16* sh  = sx + 64 * LDT;             // [64][LDT] f16 (hidden)
  _Float16* swT = sh + 64 * LDT;             // [128][LDT] f16 (W1^T then W2^T)

  const int t = threadIdx.x;
  const int wv = t >> 5, lane = t & 31;
  const int rowbase = blockIdx.x * 64;

#pragma unroll
  for (int i = 0; i < 16; ++i) {
    const int idx = i * 512 + t * 4;
    *(float4*)(sO + idx) = *(const float4*)(O + (size_t)rowbase * DD + idx);
  }
  __syncthreads();

  // LN0 (one thread per row) + stage W1^T (all threads) in parallel
  if (t < 64) {
    float* r = sO + t * DD;
    float m = 0.f;
    for (int c = 0; c < DD; ++c) m += r[c];
    m *= (1.f / DD);
    float vv = 0.f;
    for (int c = 0; c < DD; ++c) { const float d = r[c] - m; vv += d * d; }
    const float rs = rsqrtf(vv * (1.f / DD) + 1e-5f);
    for (int c = 0; c < DD; ++c) {
      const float y = (r[c] - m) * rs * g0[c] + be0[c];
      r[c] = y;
      sx[t * LDT + c] = (_Float16)y;
    }
  }
#pragma unroll
  for (int i = 0; i < 32; ++i) {
    const int idx = i * 512 + t * 4;
    const int r = idx >> 7, c = idx & 127;
    float4 w = *(const float4*)(W1 + idx);
    swT[(c + 0) * LDT + r] = (_Float16)w.x;
    swT[(c + 1) * LDT + r] = (_Float16)w.y;
    swT[(c + 2) * LDT + r] = (_Float16)w.z;
    swT[(c + 3) * LDT + r] = (_Float16)w.w;
  }
  __syncthreads();

  const int col = lane & 15;
  const int rb  = (lane < 16) ? 0 : 8;
  const int r0  = wv * 16;

  // GEMM1: sh = relu(sx @ W1 + b1)
  for (int cf = 0; cf < 8; ++cf) {
    v8f acc = {0.f,0.f,0.f,0.f,0.f,0.f,0.f,0.f};
#pragma unroll
    for (int k = 0; k < 4; ++k) {
      v16h a  = load_afrag(sx + r0 * LDT + k * 32, LDT);
      v16h bb = load_afrag(swT + (cf * 16) * LDT + k * 32, LDT);
      acc = wmma_f16(a, bb, acc);
    }
    const float bias = b1[cf * 16 + col];
#pragma unroll
    for (int v = 0; v < 8; ++v) {
      float y = acc[v] + bias;
      y = (y > 0.f) ? y : 0.f;
      sh[(r0 + rb + v) * LDT + cf * 16 + col] = (_Float16)y;
    }
  }
  __syncthreads();

#pragma unroll
  for (int i = 0; i < 32; ++i) {
    const int idx = i * 512 + t * 4;
    const int r = idx >> 7, c = idx & 127;
    float4 w = *(const float4*)(W2 + idx);
    swT[(c + 0) * LDT + r] = (_Float16)w.x;
    swT[(c + 1) * LDT + r] = (_Float16)w.y;
    swT[(c + 2) * LDT + r] = (_Float16)w.z;
    swT[(c + 3) * LDT + r] = (_Float16)w.w;
  }
  __syncthreads();

  // GEMM2 + residual (into sO, which holds LN0 output)
  for (int cf = 0; cf < 8; ++cf) {
    v8f acc = {0.f,0.f,0.f,0.f,0.f,0.f,0.f,0.f};
#pragma unroll
    for (int k = 0; k < 4; ++k) {
      v16h a  = load_afrag(sh + r0 * LDT + k * 32, LDT);
      v16h bb = load_afrag(swT + (cf * 16) * LDT + k * 32, LDT);
      acc = wmma_f16(a, bb, acc);
    }
    const float bias = b2[cf * 16 + col];
#pragma unroll
    for (int v = 0; v < 8; ++v)
      sO[(r0 + rb + v) * DD + cf * 16 + col] += acc[v] + bias;
  }
  __syncthreads();

  // LN1, normalize in place then coalesced store
  if (t < 64) {
    float* r = sO + t * DD;
    float m = 0.f;
    for (int c = 0; c < DD; ++c) m += r[c];
    m *= (1.f / DD);
    float vv = 0.f;
    for (int c = 0; c < DD; ++c) { const float d = r[c] - m; vv += d * d; }
    const float rs = rsqrtf(vv * (1.f / DD) + 1e-5f);
    for (int c = 0; c < DD; ++c) r[c] = (r[c] - m) * rs * g1[c] + be1[c];
  }
  __syncthreads();
#pragma unroll
  for (int i = 0; i < 16; ++i) {
    const int idx = i * 512 + t * 4;
    *(float4*)(out + (size_t)rowbase * DD + idx) = *(const float4*)(sO + idx);
  }
}

// ---------------------------------------------------------------------------
extern "C" void kernel_launch(void* const* d_in, const int* in_sizes, int n_in,
                              void* d_out, int out_size, void* d_ws, size_t ws_size,
                              hipStream_t stream) {
  const float* Q    = (const float*)d_in[0];
  const float* K    = (const float*)d_in[1];
  const int*   msk  = (const int*)d_in[2];
  const float* Wq   = (const float*)d_in[3];
  const float* bq   = (const float*)d_in[4];
  const float* Wk   = (const float*)d_in[5];
  const float* bk   = (const float*)d_in[6];
  const float* Wv   = (const float*)d_in[7];
  const float* bv   = (const float*)d_in[8];
  const float* Wr1  = (const float*)d_in[9];
  const float* br1  = (const float*)d_in[10];
  const float* Wr2  = (const float*)d_in[11];
  const float* br2  = (const float*)d_in[12];
  const float* g0   = (const float*)d_in[13];
  const float* be0  = (const float*)d_in[14];
  const float* g1   = (const float*)d_in[15];
  const float* be1  = (const float*)d_in[16];

  char* ws = (char*)d_ws;
  const size_t MB = 1024 * 1024;
  float*    Qpf = (float*)ws;               // 4 MB  [B,N,128] f32 (residual)
  _Float16* Qph = (_Float16*)(ws + 4 * MB); // 2 MB
  _Float16* Kph = (_Float16*)(ws + 6 * MB); // 2 MB
  _Float16* Vph = (_Float16*)(ws + 8 * MB); // 2 MB
  float*    Oh  = (float*)(ws + 10 * MB);   // 4 MB  [B,N,128] attention output

  const size_t lds1 = (size_t)2 * DD * LDT * sizeof(_Float16);         // ~68 KB
  const size_t lds2 = (size_t)64 * 32 * sizeof(int)
                    + (size_t)(2 * 32 * LDK + 4 * 16 * LDK) * sizeof(_Float16); // ~18 KB
  const size_t lds3 = (size_t)64 * DD * sizeof(float)
                    + (size_t)(2 * 64 * LDT + DD * LDT) * sizeof(_Float16);     // ~100 KB

  // Projections: 8192 rows / 128-row tiles
  proj_kernel<true><<<64, 128, lds1, stream>>>(Q, Wq, bq, Qpf, Qph);
  proj_kernel<false><<<64, 128, lds1, stream>>>(K, Wk, bk, nullptr, Kph);
  proj_kernel<false><<<64, 128, lds1, stream>>>(K, Wv, bv, nullptr, Vph);

  // Fused masked flash-attention: (N/64, B, H)
  attn_kernel<<<dim3(NN / 64, BBATCH, HH), 128, lds2, stream>>>(
      Qph, Kph, Vph, Qpf, msk, Oh);

  // LN0 + FFN + residual + LN1: 8192 rows / 64-row tiles
  epilogue_kernel<<<(BBATCH * NN) / 64, 128, lds3, stream>>>(
      Oh, Wr1, br1, Wr2, br2, g0, be0, g1, be1, (float*)d_out);
}
